// RecurrentFinSIRModel_56118042689988
// MI455X (gfx1250) — matrix operand order repaired and checked by reading, not compiled, not run on hardware
//
#include <hip/hip_runtime.h>
#include <hip/hip_bf16.h>
#include <cstdint>
#include <cstddef>

typedef __attribute__((ext_vector_type(16))) _Float16 v16h;
typedef __attribute__((ext_vector_type(8)))  float    v8f;

#define NS 4096   // stocks
#define TT 16     // timesteps
#define DD 5      // input features
#define HH 128    // hidden
#define EE 32768  // edges per graph
#define EDD 32    // edge feature dim
#define GH 512    // 4*H (gates / concat width)

__device__ __forceinline__ float sigf(float x)  { return 1.0f / (1.0f + __expf(-x)); }
__device__ __forceinline__ float leakyf(float x){ return x >= 0.0f ? x : 0.2f * x; }

// ---------------------------------------------------------------------------
// WMMA fragment loaders (contiguous k order k = 16*(lane/16) + e for A and B, wave32)
// A: 16x32 f16, row m = lane&15; lanes 0-15 hold K 0..15; lanes 16-31 K 16..31
// B: 32x16 f16 (B[k][n] = W[n][k]); col n = lane&15; lanes 0-15 K 0..15; lanes 16-31 K 16..31
// C/D: 16x16 f32; col n = lane&15; vgpr r -> row r (lanes 0-15) / 8+r (lanes 16-31)
// ---------------------------------------------------------------------------
__device__ __forceinline__ _Float16 to_h(_Float16 v) { return v; }
__device__ __forceinline__ _Float16 to_h(float v)    { return (_Float16)v; }

template <typename TA>
__device__ __forceinline__ v16h load_a_frag(const TA* __restrict__ A, int lda,
                                            int mbase, int k0, int lane) {
  const int m  = mbase + (lane & 15);
  const int kb = k0 + ((lane & 16) ? 16 : 0);
  const TA* p = A + (size_t)m * lda + kb;
  v16h a;
#pragma unroll
  for (int e = 0; e < 16; ++e) a[e] = to_h(p[e]);
  return a;
}

__device__ __forceinline__ v16h load_b_frag(const _Float16* __restrict__ W, int ldw,
                                            int nbase, int k0, int lane) {
  const int n  = nbase + (lane & 15);
  const int kb = k0 + ((lane & 16) ? 16 : 0);
  const _Float16* p = W + (size_t)n * ldw + kb;
  v16h b;
#pragma unroll
  for (int e = 0; e < 16; ++e) b[e] = p[e];
  return b;
}

// ---------------------------------------------------------------------------
// Generic GEMM:  C[m, colOff + n] = sum_k A[m,k] * W[n,k] + bias[n]
// A may be f16 (direct) or f32 (converted in-register).
// block = (32, 8) -> 8 waves, each wave one 16x16 tile; grid.x = M/16
// ---------------------------------------------------------------------------
template <typename TA>
__global__ void gemm_bias_kernel(const TA* __restrict__ A,
                                 const _Float16* __restrict__ W,
                                 const float* __restrict__ bias,
                                 float* __restrict__ C,
                                 int K, int ldc, int colOff) {
  const int lane  = threadIdx.x;
  const int wave  = threadIdx.y;
  const int mbase = blockIdx.x * 16;
  const int nbase = (blockIdx.y * 8 + wave) * 16;

  v8f acc = {};
  for (int k0 = 0; k0 < K; k0 += 32) {
    v16h a = load_a_frag(A, K, mbase, k0, lane);
    v16h b = load_b_frag(W, K, nbase, k0, lane);
    acc = __builtin_amdgcn_wmma_f32_16x16x32_f16(false, a, false, b,
                                                 (short)0, acc, false, false);
  }

  const int n    = nbase + (lane & 15);
  const float bv = bias[n];
  const int mrow = mbase + ((lane & 16) ? 8 : 0);
#pragma unroll
  for (int r = 0; r < 8; ++r) {
    C[(size_t)(mrow + r) * ldc + colOff + n] = acc[r] + bv;
  }
}

// ---------------------------------------------------------------------------
// Fused persistent LSTM: one block per 16-stock stripe, all 16 timesteps.
// 1024 threads = 32 waves; wave w computes gate n-tile w (16 of 512 columns).
// h (f16), c (f32) and the 16x512 gate tile live in LDS; gates never hit HBM.
//   gates = h @ Whh^T + sf[:,t,:] @ Wih^T + bih + bhh
//   c = sig(f)*c + sig(i)*tanh(g);  h = sig(o)*tanh(c)
// h is written to sr16 (f16, GEMM input later) and cat[:, t, 0:128] (f32).
// ---------------------------------------------------------------------------
__global__ void __launch_bounds__(1024) lstm_fused_kernel(
    const _Float16* __restrict__ whh16,
    const float* __restrict__ sf,
    const float* __restrict__ Wih,
    const float* __restrict__ bih,
    const float* __restrict__ bhh,
    _Float16* __restrict__ sr16,
    float* __restrict__ cat) {
  __shared__ _Float16 hsh[16 * HH];   // 4 KB   [row][k]
  __shared__ float    csh[16 * HH];   // 8 KB
  __shared__ float    gsh[16 * GH];   // 32 KB  [row][gate col]

  const int tid   = threadIdx.x;      // 0..1023
  const int lane  = tid & 31;
  const int wave  = tid >> 5;         // 0..31 == n-tile
  const int mbase = blockIdx.x * 16;
  const int nbase = wave * 16;

  // zero-init h, c
  for (int i = tid; i < 16 * HH; i += 1024) {
    hsh[i] = (_Float16)0.0f;
    csh[i] = 0.0f;
  }
  __syncthreads();

  // per-lane constants for this gate column
  const int n = nbase + (lane & 15);
  const float bsum = bih[n] + bhh[n];
  const float w0 = Wih[n * DD + 0], w1 = Wih[n * DD + 1], w2 = Wih[n * DD + 2];
  const float w3 = Wih[n * DD + 3], w4 = Wih[n * DD + 4];

  // preload the recurrent-weight B fragments for this n-tile (held in VGPRs)
  v16h bfr[4];
#pragma unroll
  for (int kk = 0; kk < 4; ++kk)
    bfr[kk] = load_b_frag(whh16, HH, nbase, kk * 32, lane);

  const int rrow = (lane & 16) ? 8 : 0;

  for (int t = 0; t < TT; ++t) {
    // gates tile: (16 x 512) = h(LDS) @ Whh^T
    v8f acc = {};
#pragma unroll
    for (int kk = 0; kk < 4; ++kk) {
      v16h a = load_a_frag(hsh, HH, 0, kk * 32, lane);   // ds reads
      acc = __builtin_amdgcn_wmma_f32_16x16x32_f16(false, a, false, bfr[kk],
                                                   (short)0, acc, false, false);
    }
    // epilogue: add input projection, write gate tile to LDS
#pragma unroll
    for (int r = 0; r < 8; ++r) {
      const int row = rrow + r;
      const float* s = sf + ((size_t)(mbase + row) * TT + t) * DD;
      gsh[row * GH + n] =
          acc[r] + bsum + s[0] * w0 + s[1] * w1 + s[2] * w2 + s[3] * w3 + s[4] * w4;
    }
    __syncthreads();

    // cell update: 2048 elements over 1024 threads
    for (int i = tid; i < 16 * HH; i += 1024) {
      const int row = i >> 7, j = i & (HH - 1);
      const float* g = gsh + row * GH;
      const float gi = sigf(g[j]);
      const float gf = sigf(g[HH + j]);
      const float gg = tanhf(g[2 * HH + j]);
      const float go = sigf(g[3 * HH + j]);
      const float cv = gf * csh[i] + gi * gg;
      csh[i] = cv;
      const float h = go * tanhf(cv);
      hsh[row * HH + j] = (_Float16)h;
      const size_t m = (size_t)(mbase + row);
      sr16[(m * TT + t) * HH + j] = (_Float16)h;
      cat[(m * TT + t) * GH + j] = h;
    }
    __syncthreads();
  }
}

// Normalize: sf[n,t,d] = x[n,t,d] / mean_t(x[n,:,d])
__global__ void norm_kernel(const float* __restrict__ x, float* __restrict__ sf) {
  const int idx = blockIdx.x * blockDim.x + threadIdx.x;   // NS*DD
  if (idx >= NS * DD) return;
  const int n = idx / DD, d = idx % DD;
  float s = 0.0f;
#pragma unroll
  for (int t = 0; t < TT; ++t) s += x[((size_t)n * TT + t) * DD + d];
  const float inv = (float)TT / s;
#pragma unroll
  for (int t = 0; t < TT; ++t) {
    const size_t o = ((size_t)n * TT + t) * DD + d;
    sf[o] = x[o] * inv;
  }
}

__global__ void cvt_f16_kernel(const float* __restrict__ in, _Float16* __restrict__ out, int n) {
  for (int i = blockIdx.x * blockDim.x + threadIdx.x; i < n; i += gridDim.x * blockDim.x)
    out[i] = (_Float16)in[i];
}

__global__ void zero_f32_kernel(float* __restrict__ p, int n) {
  for (int i = blockIdx.x * blockDim.x + threadIdx.x; i < n; i += gridDim.x * blockDim.x)
    p[i] = 0.0f;
}

// Fused edge message + segment-sum: ft[dst] += leaky(eq[dst] + ek[src] + e)
// one block per edge, 256 threads over T*H/4 = 512 float4s
__global__ void edge_msg_kernel(const int* __restrict__ src,
                                const int* __restrict__ dst,
                                const float* __restrict__ eq,
                                const float* __restrict__ ek,
                                const float* __restrict__ eed,
                                float* __restrict__ ft) {
  const int e = blockIdx.x;
  const int s = src[e], d = dst[e];
  const float4* eq4 = (const float4*)(eq + (size_t)d * (TT * HH));
  const float4* ek4 = (const float4*)(ek + (size_t)s * (TT * HH));
  const float4* er4 = (const float4*)(eed + (size_t)e * HH);
  float* ftr = ft + (size_t)d * (TT * HH);
  for (int idx = threadIdx.x; idx < (TT * HH) / 4; idx += blockDim.x) {
    const float4 a = eq4[idx];
    const float4 b = ek4[idx];
    const float4 c = er4[idx & (HH / 4 - 1)];
    float* o = ftr + idx * 4;
    atomicAdd(o + 0, leakyf(a.x + b.x + c.x));
    atomicAdd(o + 1, leakyf(a.y + b.y + c.y));
    atomicAdd(o + 2, leakyf(a.z + b.z + c.z));
    atomicAdd(o + 3, leakyf(a.w + b.w + c.w));
  }
}

// Attention pooling + MLP head; one block (256 threads) per stock
__global__ void head_kernel(const float* __restrict__ cat,
                            const float* __restrict__ attnW,
                            const float* __restrict__ attnB,
                            const float* __restrict__ W1,
                            const float* __restrict__ b1,
                            const float* __restrict__ W2,
                            const float* __restrict__ b2,
                            const float* __restrict__ sf,
                            float* __restrict__ out) {
  __shared__ float scores[TT];
  __shared__ float aw[TT];
  __shared__ float pooled[GH];
  __shared__ float red[256];
  __shared__ float hbuf[HH];

  const int n = blockIdx.x, tid = threadIdx.x;
  const float* base = cat + (size_t)n * TT * GH;

  for (int t = 0; t < TT; ++t) {
    const float* row = base + (size_t)t * GH;
    red[tid] = row[tid] * attnW[tid] + row[tid + 256] * attnW[tid + 256];
    __syncthreads();
    for (int off = 128; off > 0; off >>= 1) {
      if (tid < off) red[tid] += red[tid + off];
      __syncthreads();
    }
    if (tid == 0) scores[t] = red[0] + attnB[0];
    __syncthreads();
  }

  if (tid == 0) {
    float mx = scores[0];
#pragma unroll
    for (int t = 1; t < TT; ++t) mx = fmaxf(mx, scores[t]);
    float sum = 0.0f;
#pragma unroll
    for (int t = 0; t < TT; ++t) { aw[t] = __expf(scores[t] - mx); sum += aw[t]; }
    const float inv = 1.0f / sum;
#pragma unroll
    for (int t = 0; t < TT; ++t) aw[t] *= inv;
  }
  __syncthreads();

  float p0 = 0.0f, p1 = 0.0f;
#pragma unroll
  for (int t = 0; t < TT; ++t) {
    const float* row = base + (size_t)t * GH;
    p0 += aw[t] * row[tid];
    p1 += aw[t] * row[tid + 256];
  }
  pooled[tid] = p0; pooled[tid + 256] = p1;
  __syncthreads();

  if (tid < HH) {
    float acc = b1[tid];
    const float* w = W1 + (size_t)tid * GH;
    for (int j = 0; j < GH; ++j) acc += pooled[j] * w[j];
    hbuf[tid] = leakyf(acc);
  }
  __syncthreads();

  red[tid] = (tid < HH) ? hbuf[tid] * W2[tid] : 0.0f;
  __syncthreads();
  for (int off = 128; off > 0; off >>= 1) {
    if (tid < off) red[tid] += red[tid + off];
    __syncthreads();
  }
  if (tid == 0) {
    const float o = leakyf(red[0] + b2[0]);
    const float sl = sf[((size_t)n * TT + (TT - 1)) * DD + 0];
    out[n] = o / sl - 1.0f;
  }
}

// ---------------------------------------------------------------------------
extern "C" void kernel_launch(void* const* d_in, const int* in_sizes, int n_in,
                              void* d_out, int out_size, void* d_ws, size_t ws_size,
                              hipStream_t stream) {
  (void)in_sizes; (void)n_in; (void)out_size; (void)ws_size;

  const float* x     = (const float*)d_in[0];
  const float* Wih   = (const float*)d_in[34];
  const float* Whh   = (const float*)d_in[35];
  const float* bih   = (const float*)d_in[36];
  const float* bhh   = (const float*)d_in[37];
  const float* attnW = (const float*)d_in[38];
  const float* attnB = (const float*)d_in[39];
  const float* W1 = (const float*)d_in[40];
  const float* b1 = (const float*)d_in[41];
  const float* W2 = (const float*)d_in[42];
  const float* b2 = (const float*)d_in[43];

  // workspace carve (256B aligned)
  char* p = (char*)d_ws;
  auto take = [&](size_t bytes) {
    char* r = p;
    p += (bytes + 255) & ~(size_t)255;
    return r;
  };
  float*    sf    = (float*)take((size_t)NS * TT * DD * 4);
  _Float16* sr16  = (_Float16*)take((size_t)NS * TT * HH * 2);
  float*    eq    = (float*)take((size_t)NS * TT * HH * 4);
  float*    ek    = (float*)take((size_t)NS * TT * HH * 4);
  float*    eed   = (float*)take((size_t)EE * HH * 4);
  float*    ft    = (float*)take((size_t)NS * TT * HH * 4);
  float*    cat   = (float*)take((size_t)NS * TT * GH * 4);
  _Float16* whh16 = (_Float16*)take((size_t)GH * HH * 2);
  _Float16* wq16  = (_Float16*)take((size_t)HH * HH * 2);
  _Float16* wk16  = (_Float16*)take((size_t)HH * HH * 2);
  _Float16* we16  = (_Float16*)take((size_t)HH * EDD * 2);
  _Float16* wr16  = (_Float16*)take((size_t)HH * HH * 2);

  const dim3 gblk(32, 8);

  // 1) normalize
  norm_kernel<<<(NS * DD + 255) / 256, 256, 0, stream>>>(x, sf);

  // 2) fused LSTM (single launch, gates/h/c in LDS)
  cvt_f16_kernel<<<(GH * HH + 255) / 256, 256, 0, stream>>>(Whh, whh16, GH * HH);
  lstm_fused_kernel<<<NS / 16, 1024, 0, stream>>>(whh16, sf, Wih, bih, bhh, sr16, cat);

  // 3) graph layers
  for (int g = 0; g < 3; ++g) {
    const int base = 1 + g * 11;
    const int*   src   = (const int*)d_in[base + 0];
    const int*   dst   = (const int*)d_in[base + 1];
    const float* efeat = (const float*)d_in[base + 2];
    const float* Wq = (const float*)d_in[base + 3];
    const float* bq = (const float*)d_in[base + 4];
    const float* Wk = (const float*)d_in[base + 5];
    const float* bk = (const float*)d_in[base + 6];
    const float* We = (const float*)d_in[base + 7];
    const float* be = (const float*)d_in[base + 8];
    const float* Wr = (const float*)d_in[base + 9];
    const float* br = (const float*)d_in[base + 10];

    cvt_f16_kernel<<<(HH * HH + 255) / 256, 256, 0, stream>>>(Wq, wq16, HH * HH);
    cvt_f16_kernel<<<(HH * HH + 255) / 256, 256, 0, stream>>>(Wk, wk16, HH * HH);
    cvt_f16_kernel<<<(HH * EDD + 255) / 256, 256, 0, stream>>>(We, we16, HH * EDD);
    cvt_f16_kernel<<<(HH * HH + 255) / 256, 256, 0, stream>>>(Wr, wr16, HH * HH);

    // e = efeat @ We^T + be   [E, 128], K = 32 (f32 A converted in-register)
    gemm_bias_kernel<float><<<dim3(EE / 16, 1), gblk, 0, stream>>>(
        efeat, we16, be, eed, EDD, HH, 0);
    // eq = sr @ Wq^T + bq ; ek = sr @ Wk^T + bk   [N*T, 128], K = 128
    gemm_bias_kernel<_Float16><<<dim3((NS * TT) / 16, 1), gblk, 0, stream>>>(
        sr16, wq16, bq, eq, HH, HH, 0);
    gemm_bias_kernel<_Float16><<<dim3((NS * TT) / 16, 1), gblk, 0, stream>>>(
        sr16, wk16, bk, ek, HH, HH, 0);

    // fused message + segment-sum into ft
    zero_f32_kernel<<<4096, 256, 0, stream>>>(ft, NS * TT * HH);
    edge_msg_kernel<<<EE, 256, 0, stream>>>(src, dst, eq, ek, eed, ft);

    // out_g = ft @ Wr^T + br  -> cat columns [(1+g)*128, (2+g)*128)  (f32 A)
    gemm_bias_kernel<float><<<dim3((NS * TT) / 16, 1), gblk, 0, stream>>>(
        ft, wr16, br, cat, HH, GH, (1 + g) * HH);
  }

  // 4) attention pool + MLP head
  head_kernel<<<NS, 256, 0, stream>>>(cat, attnW, attnB, W1, b1, W2, b2, sf, (float*)d_out);
}
